// LongformerAttention_29360396435952
// MI455X (gfx1250) — matrix operand order repaired and checked by reading, not compiled, not run on hardware
//
#include <hip/hip_runtime.h>
#include <hip/hip_bf16.h>

typedef __attribute__((ext_vector_type(16))) _Float16 v16h;
typedef __attribute__((ext_vector_type(8)))  _Float16 v8h;
typedef __attribute__((ext_vector_type(8)))  float    v8f;

#define LF_H   12
#define LF_S   4096
#define LF_D   768
#define LF_DH  64
#define LF_W   256
#define LF_NB  16
#define LF_NEG (-10000.0f)

// LDS layout for lf_attn (dynamic shared), bytes:
//   kctx  [768][64]  f16  @      0  (96 KB)   K context, [key_local][dh]
//   vctx  [64][768]  f16  @  98304  (96 KB)   V context, [dh][key_local]
//   pbuf  16 x 512   f16  @ 196608  (16 KB)   per-wave P transpose slabs
#define LF_LDS_K   0
#define LF_LDS_V   98304
#define LF_LDS_P   196608
#define LF_LDS_SZ  212992

static __device__ inline v16h join16(v8h lo, v8h hi) {
  return __builtin_shufflevector(lo, hi, 0,1,2,3,4,5,6,7,8,9,10,11,12,13,14,15);
}

// load 16 contiguous f16 (32B, 16B aligned)
static __device__ inline v16h ld16h(const _Float16* p) {
  v8h lo = *(const v8h*)(p);
  v8h hi = *(const v8h*)(p + 8);
  return join16(lo, hi);
}

static __device__ inline v8f wmma_f16(v16h a, v16h b, v8f c) {
  // (neg_a, A, neg_b, B, c_mod, C, reuse_a, reuse_b)
  return __builtin_amdgcn_wmma_f32_16x16x32_f16(false, a, false, b, (short)0, c,
                                                false, false);
}

// Async global -> LDS copy, 16 bytes per lane (ASYNCcnt-tracked, CDNA5 TDM-lite)
static __device__ inline void async_cp16(unsigned lds_off, const void* gptr) {
  asm volatile("global_load_async_to_lds_b128 %0, %1, off"
               :: "v"(lds_off), "v"(gptr)
               : "memory");
}
static __device__ inline void wait_async0() {
  asm volatile("s_wait_asynccnt 0x0" ::: "memory");
}

// ---------------------------------------------------------------------------
// Kernel A: fused QKV projection.
// X[B*S, 768] @ W[768,768] + b, per head strip of 64 output cols.
//   mat 0 -> Qf16 [B,H,S,64] (prescaled by 1/8)
//   mat 1 -> Kf16 [B,H,S,64]
//   mat 2 -> Vt16 [B,H,64,S] (transposed so PV B-operand is contiguous)
// ---------------------------------------------------------------------------
__global__ __launch_bounds__(256) void lf_qkv_proj(
    const float* __restrict__ X,
    const float* __restrict__ Wq, const float* __restrict__ bq,
    const float* __restrict__ Wk, const float* __restrict__ bk,
    const float* __restrict__ Wv, const float* __restrict__ bv,
    _Float16* __restrict__ Qf, _Float16* __restrict__ Kf,
    _Float16* __restrict__ Vt)
{
  __shared__ __align__(16) _Float16 wt[64 * 40];  // [o_local][k_local], 80B rows

  const int mat = blockIdx.z;
  const float* W    = (mat == 0) ? Wq : ((mat == 1) ? Wk : Wv);
  const float* bias = (mat == 0) ? bq : ((mat == 1) ? bk : bv);
  const int h     = blockIdx.y;
  const int obase = h * LF_DH;
  const int tid   = threadIdx.x;
  const int wave  = tid >> 5;
  const int lane  = tid & 31;
  const int col   = lane & 15;
  const int hs    = lane >> 4;
  const int m0    = blockIdx.x * 128 + wave * 16;
  const int arow  = m0 + col;

  v8f zero = {};
  v8f acc[4];
  #pragma unroll
  for (int t = 0; t < 4; ++t) acc[t] = zero;

  for (int k = 0; k < LF_D; k += 32) {
    {
      const int r  = tid >> 3;
      const int c0 = (tid & 7) * 8;
      const float* wrow = W + (size_t)(k + r) * LF_D + obase + c0;
      #pragma unroll
      for (int j = 0; j < 8; ++j)
        wt[(c0 + j) * 40 + r] = (_Float16)wrow[j];
    }
    __syncthreads();

    v16h a;
    {
      const float* xr = X + (size_t)arow * LF_D + k + hs * 16;
      #pragma unroll
      for (int j = 0; j < 8; ++j) {
        a[j]     = (_Float16)xr[j];
        a[8 + j] = (_Float16)xr[8 + j];
      }
    }

    #pragma unroll
    for (int t = 0; t < 4; ++t) {
      const _Float16* bp = &wt[(t * 16 + col) * 40 + hs * 16];
      v16h bf = ld16h(bp);
      acc[t] = wmma_f16(a, bf, acc[t]);
    }
    __syncthreads();
  }

  const int bidx  = m0 / LF_S;
  const int srow0 = m0 - bidx * LF_S;
  const size_t bh = (size_t)(bidx * LF_H + h);

  #pragma unroll
  for (int t = 0; t < 4; ++t) {
    const int dh = t * 16 + col;
    const float bv_ = bias[obase + dh];
    #pragma unroll
    for (int r = 0; r < 8; ++r) {
      const int srow = srow0 + r + 8 * hs;
      float val = acc[t][r] + bv_;
      if (mat == 0) {
        val *= 0.125f;
        Qf[(bh * LF_S + srow) * LF_DH + dh] = (_Float16)val;
      } else if (mat == 1) {
        Kf[(bh * LF_S + srow) * LF_DH + dh] = (_Float16)val;
      } else {
        Vt[(bh * LF_DH + dh) * LF_S + srow] = (_Float16)val;
      }
    }
  }
}

// ---------------------------------------------------------------------------
// Kernel B: banded attention with online softmax.
// Grid (nblk, H, B); 512 threads = 16 waves; wave qt owns queries
// [i*256 + qt*16, +16). K/V 3-block context is async-staged into LDS once
// per workgroup; all WMMA B-operands then come from LDS.
// ---------------------------------------------------------------------------
__global__ __launch_bounds__(512) void lf_attn(
    const _Float16* __restrict__ Qf,
    const _Float16* __restrict__ Kf,
    const _Float16* __restrict__ Vt,
    const unsigned char* __restrict__ maskp,
    float* __restrict__ out)
{
  extern __shared__ __align__(16) char smem[];
  _Float16* kctx = (_Float16*)(smem + LF_LDS_K);
  _Float16* vctx = (_Float16*)(smem + LF_LDS_V);

  const int i    = blockIdx.x;
  const int h    = blockIdx.y;
  const int b    = blockIdx.z;
  const int tid  = threadIdx.x;
  const int wave = tid >> 5;
  const int lane = tid & 31;
  const int col  = lane & 15;
  const int g    = lane >> 4;
  const int qt   = wave;
  const int s0q  = i * LF_W + qt * 16;
  const size_t bh = (size_t)(b * LF_H + h);

  const _Float16* Qb = Qf + bh * LF_S * LF_DH;
  const _Float16* Kb = Kf + bh * LF_S * LF_DH;
  const _Float16* Vb = Vt + bh * LF_DH * LF_S;
  _Float16* pw = (_Float16*)(smem + LF_LDS_P) + wave * 512;

  // ---- async-stage valid K/V context into LDS (16B chunks, 512 threads) ----
  {
    const int klo = (i == 0) ? LF_W : 0;            // valid key_local range
    const int khi = (i == LF_NB - 1) ? 2 * LF_W : 3 * LF_W;
    const int kglo = (i - 1) * LF_W + klo;          // >= 0

    // K: contiguous [klo,khi) rows of 64 f16 => (khi-klo)*128 bytes
    {
      const int bytes = (khi - klo) * LF_DH * 2;
      const char* gsrc = (const char*)(Kb + (size_t)kglo * LF_DH);
      const unsigned ldst = (unsigned)(uintptr_t)(kctx + (size_t)klo * LF_DH);
      for (int ofs = tid * 16; ofs < bytes; ofs += 512 * 16)
        async_cp16(ldst + ofs, gsrc + ofs);
    }
    // V: 64 rows (dh), each a contiguous (khi-klo)*2 byte segment
    {
      const int segb   = (khi - klo) * 2;
      const int segc   = segb >> 4;                 // 16B chunks per row
      const int total  = LF_DH * segc;
      for (int c = tid; c < total; c += 512) {
        const int row = c / segc;
        const int ofs = (c - row * segc) * 16;
        const char* gsrc = (const char*)(Vb + (size_t)row * LF_S + kglo) + ofs;
        const unsigned ldst =
            (unsigned)(uintptr_t)(vctx + (size_t)row * (3 * LF_W) + klo) + ofs;
        async_cp16(ldst, gsrc);
      }
    }
    wait_async0();
    __syncthreads();
  }

  // ---- Q fragments (dh 0..31 and 32..63), resident for the whole wave ----
  v16h qa0, qa1;
  {
    const _Float16* qp = Qb + (size_t)(s0q + col) * LF_DH + g * 16;
    #pragma unroll
    for (int j = 0; j < 8; ++j) {
      qa0[j]     = qp[j];        qa0[8 + j] = qp[8 + j];
      qa1[j]     = qp[32 + j];   qa1[8 + j] = qp[40 + j];
    }
  }

  float mstat[8], lstat[8];
  v8f zero = {};
  v8f Oacc[4];
  #pragma unroll
  for (int r = 0; r < 8; ++r) { mstat[r] = -3.0e38f; lstat[r] = 0.0f; }
  #pragma unroll
  for (int t = 0; t < 4; ++t) Oacc[t] = zero;

  // chunk range: skip edge-padding blocks and keys outside the band
  int clo = qt >> 1;
  if (i == 0 && clo < 8) clo = 8;
  int chi = (i == LF_NB - 1) ? 15 : 23;
  {
    int chiband = (16 * qt + 527) >> 5;
    if (chiband < chi) chi = chiband;
  }

  for (int c = clo; c <= chi; ++c) {
    const int kl0 = c * 32;
    const int kg0 = (i - 1) * LF_W + kl0;

    // ---- scores: two 16x16 tiles over 32 keys, reduction over Dh=64 ----
    v8f s0 = zero, s1 = zero;
    {
      const _Float16* kp0 = kctx + (size_t)(kl0 + col) * LF_DH + g * 16;
      const _Float16* kp1 = kp0 + 16 * LF_DH;
      v16h kb;
      kb = ld16h(kp0);       s0 = wmma_f16(qa0, kb, s0);
      kb = ld16h(kp0 + 32);  s0 = wmma_f16(qa1, kb, s0);
      kb = ld16h(kp1);       s1 = wmma_f16(qa0, kb, s1);
      kb = ld16h(kp1 + 32);  s1 = wmma_f16(qa1, kb, s1);
    }

    // per-key bias (padding edges + key mask), one per lane per tile
    const int kl_a = kl0 + col,      kl_b = kl0 + 16 + col;
    const int kg_a = kg0 + col,      kg_b = kg0 + 16 + col;
    const float bias_a =
        (kg_a >= 0 && kg_a < LF_S && maskp[(size_t)b * LF_S + kg_a]) ? 0.0f : LF_NEG;
    const float bias_b =
        (kg_b >= 0 && kg_b < LF_S && maskp[(size_t)b * LF_S + kg_b]) ? 0.0f : LF_NEG;

    // ---- band mask + online softmax (row stats across lanes via shfl) ----
    float p0[8], p1[8], scale[8];
    #pragma unroll
    for (int r = 0; r < 8; ++r) {
      const int qrow = qt * 16 + r + 8 * g;
      float sa = ((kl_a >= qrow) && (kl_a <= qrow + 2 * LF_W)) ? (s0[r] + bias_a)
                                                               : LF_NEG;
      float sb = ((kl_b >= qrow) && (kl_b <= qrow + 2 * LF_W)) ? (s1[r] + bias_b)
                                                               : LF_NEG;
      float tm = fmaxf(sa, sb);
      tm = fmaxf(tm, __shfl_xor(tm, 1, 32));
      tm = fmaxf(tm, __shfl_xor(tm, 2, 32));
      tm = fmaxf(tm, __shfl_xor(tm, 4, 32));
      tm = fmaxf(tm, __shfl_xor(tm, 8, 32));
      const float mo = mstat[r];
      const float mn = fmaxf(mo, tm);
      scale[r] = __expf(mo - mn);
      const float pa = __expf(sa - mn);
      const float pb = __expf(sb - mn);
      p0[r] = pa; p1[r] = pb;
      float rs = pa + pb;
      rs += __shfl_xor(rs, 1, 32);
      rs += __shfl_xor(rs, 2, 32);
      rs += __shfl_xor(rs, 4, 32);
      rs += __shfl_xor(rs, 8, 32);
      lstat[r] = lstat[r] * scale[r] + rs;
      mstat[r] = mn;
    }

    // ---- C-layout P -> LDS (row-major 16x32) -> A-layout fragment ----
    #pragma unroll
    for (int r = 0; r < 8; ++r) {
      const int row = r + 8 * g;
      pw[row * 32 + col]      = (_Float16)p0[r];
      pw[row * 32 + 16 + col] = (_Float16)p1[r];
    }
    v16h pa;
    {
      const _Float16* pp = pw + (size_t)col * 32 + g * 16;
      #pragma unroll
      for (int j = 0; j < 8; ++j) { pa[j] = pp[j]; pa[8 + j] = pp[8 + j]; }
    }

    // ---- O = O*scale + P @ V  (4 dh tiles; vctx is [dh][key] contiguous) ----
    #pragma unroll
    for (int t = 0; t < 4; ++t) {
      v8f o = Oacc[t];
      #pragma unroll
      for (int r = 0; r < 8; ++r) o[r] *= scale[r];
      const _Float16* vp = vctx + (size_t)(t * 16 + col) * (3 * LF_W) + kl0 + g * 16;
      v16h vbf = ld16h(vp);
      Oacc[t] = wmma_f16(pa, vbf, o);
    }
  }

  // ---- finalize: divide by l, zero masked-query rows, store [B,S,H*Dh] ----
  float fin[8];
  #pragma unroll
  for (int r = 0; r < 8; ++r) {
    const int sq = s0q + r + 8 * g;
    const float qv = maskp[(size_t)b * LF_S + sq] ? 1.0f : 0.0f;
    fin[r] = qv / fmaxf(lstat[r], 1e-30f);
  }
  #pragma unroll
  for (int t = 0; t < 4; ++t) {
    #pragma unroll
    for (int r = 0; r < 8; ++r) {
      const int sq = s0q + r + 8 * g;
      out[((size_t)(b * LF_S + sq)) * LF_D + h * LF_DH + t * 16 + col] =
          Oacc[t][r] * fin[r];
    }
  }
}

extern "C" void kernel_launch(void* const* d_in, const int* in_sizes, int n_in,
                              void* d_out, int out_size, void* d_ws, size_t ws_size,
                              hipStream_t stream) {
  (void)in_sizes; (void)n_in; (void)out_size; (void)ws_size;
  const float* X            = (const float*)d_in[0];
  // d_in[1], d_in[2] (key/value tensors) are ignored by the reference.
  const unsigned char* mask = (const unsigned char*)d_in[3];
  const float* Wq = (const float*)d_in[4];
  const float* bq = (const float*)d_in[5];
  const float* Wk = (const float*)d_in[6];
  const float* bk = (const float*)d_in[7];
  const float* Wv = (const float*)d_in[8];
  const float* bv = (const float*)d_in[9];
  float* out = (float*)d_out;

  const size_t qElems = (size_t)2 * LF_H * LF_S * LF_DH;
  _Float16* Qf = (_Float16*)d_ws;
  _Float16* Kf = Qf + qElems;
  _Float16* Vt = Kf + qElems;

  // allow >64KB dynamic LDS (WGP has 320KB on CDNA5)
  (void)hipFuncSetAttribute((const void*)lf_attn,
                            hipFuncAttributeMaxDynamicSharedMemorySize,
                            LF_LDS_SZ);

  dim3 gA(64, LF_H, 3);
  lf_qkv_proj<<<gA, dim3(256), 0, stream>>>(X, Wq, bq, Wk, bk, Wv, bv, Qf, Kf, Vt);

  dim3 gB(LF_NB, LF_H, 2);
  lf_attn<<<gB, dim3(512), LF_LDS_SZ, stream>>>(Qf, Kf, Vt, mask, out);
}
